// Graph_RNN_44925357916527
// MI455X (gfx1250) — compile-verified, matrix operand order checked by reading, not run on hardware
//
#include <hip/hip_runtime.h>
#include <hip/hip_bf16.h>
#include <math.h>

#define NBB   2
#define NT    8
#define NNODE 50000
#define NF    32
#define NH    64
#define NEDGE 800000
#define TOUT  9          // NT + pred_hor(1)
#define WAVES 4          // waves per node-kernel block
#define SSTR  68         // LDS scratch row stride (floats), conflict-avoiding

typedef __attribute__((ext_vector_type(16))) __bf16 v16bf;
typedef __attribute__((ext_vector_type(8)))  __bf16 v8bf;
typedef __attribute__((ext_vector_type(8)))  float  v8f;

// ---------------------------------------------------------------------------
// WMMA wrapper: D(f32 16x16) = A(bf16 16x32) x B(bf16 32x16) + C
// ---------------------------------------------------------------------------
__device__ __forceinline__ v8f wmma_bf16(v16bf a, v16bf b, v8f c) {
  return __builtin_amdgcn_wmma_f32_16x16x32_bf16(
      /*neg_a=*/false, a, /*neg_b=*/false, b,
      /*c_mod=*/(short)0, c, /*reuse_a=*/false, /*reuse_b=*/false);
}

// A-fragment (16x32 bf16) from row-major f32 global memory.
// ISA 16-bit A layout: lane(m,hi) holds K in [8*hi,8*hi+8) and [16+8*hi,16+8*hi+8).
__device__ __forceinline__ v16bf afrag_g(const float* __restrict__ base, int ld,
                                         int kbase, int m, int hi) {
  const float* r = base + m * ld + kbase + hi * 8;
  float4 a0 = *(const float4*)(r);
  float4 a1 = *(const float4*)(r + 4);
  float4 a2 = *(const float4*)(r + 16);
  float4 a3 = *(const float4*)(r + 20);
  v16bf v;
  v[0]=(__bf16)a0.x; v[1]=(__bf16)a0.y; v[2]=(__bf16)a0.z; v[3]=(__bf16)a0.w;
  v[4]=(__bf16)a1.x; v[5]=(__bf16)a1.y; v[6]=(__bf16)a1.z; v[7]=(__bf16)a1.w;
  v[8]=(__bf16)a2.x; v[9]=(__bf16)a2.y; v[10]=(__bf16)a2.z; v[11]=(__bf16)a2.w;
  v[12]=(__bf16)a3.x; v[13]=(__bf16)a3.y; v[14]=(__bf16)a3.z; v[15]=(__bf16)a3.w;
  return v;
}

// A-fragment from per-wave LDS f32 scratch (row stride SSTR).
__device__ __forceinline__ v16bf afrag_s(const float* S, int kbase, int m, int hi) {
  const float* r = S + m * SSTR + kbase + hi * 8;
  float4 a0 = *(const float4*)(r);
  float4 a1 = *(const float4*)(r + 4);
  float4 a2 = *(const float4*)(r + 16);
  float4 a3 = *(const float4*)(r + 20);
  v16bf v;
  v[0]=(__bf16)a0.x; v[1]=(__bf16)a0.y; v[2]=(__bf16)a0.z; v[3]=(__bf16)a0.w;
  v[4]=(__bf16)a1.x; v[5]=(__bf16)a1.y; v[6]=(__bf16)a1.z; v[7]=(__bf16)a1.w;
  v[8]=(__bf16)a2.x; v[9]=(__bf16)a2.y; v[10]=(__bf16)a2.z; v[11]=(__bf16)a2.w;
  v[12]=(__bf16)a3.x; v[13]=(__bf16)a3.y; v[14]=(__bf16)a3.z; v[15]=(__bf16)a3.w;
  return v;
}

// B-fragment (32x16 bf16), B[k,n] = W[nbase+n, kbase+k] for weight W (out x in,
// row-major, pre-converted bf16 in LDS). Two ds_load_b128 per lane.
__device__ __forceinline__ v16bf bfrag_s(const __bf16* W, int ld, int nbase,
                                         int kbase, int n, int hi) {
  const __bf16* r = W + (nbase + n) * ld + kbase + hi * 8;
  v8bf b0 = *(const v8bf*)(r);
  v8bf b1 = *(const v8bf*)(r + 16);
  v16bf v;
#pragma unroll
  for (int j = 0; j < 8; ++j) { v[j] = b0[j]; v[8 + j] = b1[j]; }
  return v;
}

// ---------------------------------------------------------------------------
// H0 broadcast and agg zeroing
// ---------------------------------------------------------------------------
__global__ void fill_H0(const float* __restrict__ initH, float* __restrict__ H,
                        size_t n) {
  size_t i = (size_t)blockIdx.x * blockDim.x + threadIdx.x;
  if (i < n) H[i] = initH[i & (NH - 1)];
}

__global__ void zero_buf(float* __restrict__ p, size_t n) {
  size_t i = (size_t)blockIdx.x * blockDim.x + threadIdx.x;
  if (i < n) p[i] = 0.0f;
}

// ---------------------------------------------------------------------------
// Edge gather/scatter: agg[b,dst,:] += H[b,src,:] * ef[b,t,e]
// One wave per (b,e); lane handles 2 channels. H/agg are L2-resident (25.6 MB).
// ---------------------------------------------------------------------------
__global__ void edge_scatter(const float* __restrict__ H,
                             const int* __restrict__ esrc,
                             const int* __restrict__ edst,
                             const float* __restrict__ efeat, int ef_t,
                             float* __restrict__ agg) {
  long long gid = (long long)blockIdx.x * blockDim.x + threadIdx.x;
  long long w = gid >> 5;
  if (w >= (long long)NBB * NEDGE) return;
  int lane = threadIdx.x & 31;
  int b = (int)(w / NEDGE);
  int e = (int)(w % NEDGE);
  int src = esrc[e];
  int dst = edst[e];
  float f = efeat[((size_t)b * NT + ef_t) * NEDGE + e];
  float2 hv = *(const float2*)(H + ((size_t)b * NNODE + src) * NH + lane * 2);
  float* d = agg + ((size_t)b * NNODE + dst) * NH + lane * 2;
  atomicAdd(d, hv.x * f);
  atomicAdd(d + 1, hv.y * f);
}

// ---------------------------------------------------------------------------
// Node update: one wave per 16-node tile.
//   pre  = H@A^T + x@Bm^T + agg@C^T + D      (WMMA bf16, f32 acc)
//   Hnew = H + tanh(pre)                      (in place)
//   out  = relu(relu(Hnew@W1^T+b1)@W2^T+b2)@W3^T+b3 + x@Fm^T + G
// ---------------------------------------------------------------------------
__global__ __launch_bounds__(WAVES * 32) void node_step(
    const float* __restrict__ H, const float* __restrict__ agg,
    const float* __restrict__ xbase, int x_tstride, int x_t,
    float* __restrict__ out, int out_t,
    const float* __restrict__ A, const float* __restrict__ Bm,
    const float* __restrict__ C, const float* __restrict__ Dv,
    const float* __restrict__ Fm, const float* __restrict__ G,
    const float* __restrict__ W1, const float* __restrict__ b1,
    const float* __restrict__ W2, const float* __restrict__ b2,
    const float* __restrict__ W3, const float* __restrict__ b3) {
  __shared__ __align__(16) __bf16 sA[NH * NH];
  __shared__ __align__(16) __bf16 sC[NH * NH];
  __shared__ __align__(16) __bf16 sW1[NH * NH];
  __shared__ __align__(16) __bf16 sW2[NH * NH];
  __shared__ __align__(16) __bf16 sW3[NF * NH];
  __shared__ __align__(16) __bf16 sBm[NH * NF];
  __shared__ __align__(16) __bf16 sFm[NF * NF];
  __shared__ float sD[NH], sb1[NH], sb2[NH], sb3[NF], sG[NF];
  __shared__ __align__(16) float sScr[WAVES * 16 * SSTR];

  const int tid = threadIdx.x;
  for (int i = tid; i < NH * NH; i += WAVES * 32) {
    sA[i] = (__bf16)A[i];  sC[i] = (__bf16)C[i];
    sW1[i] = (__bf16)W1[i]; sW2[i] = (__bf16)W2[i];
  }
  for (int i = tid; i < NF * NH; i += WAVES * 32) {
    sW3[i] = (__bf16)W3[i]; sBm[i] = (__bf16)Bm[i];
  }
  for (int i = tid; i < NF * NF; i += WAVES * 32) sFm[i] = (__bf16)Fm[i];
  if (tid < NH) { sD[tid] = Dv[tid]; sb1[tid] = b1[tid]; sb2[tid] = b2[tid]; }
  if (tid < NF) { sb3[tid] = b3[tid]; sG[tid] = G[tid]; }
  __syncthreads();

  const int wave = tid >> 5;
  const int lane = tid & 31;
  const int tile = blockIdx.x * WAVES + wave;
  const int tilesPerB = NNODE / 16;
  if (tile >= NBB * tilesPerB) return;   // whole-wave exit: EXEC stays all-1s

  const int b = tile / tilesPerB;
  const int nb = (tile % tilesPerB) * 16;
  const int m = lane & 15;   // A-frag row / B-frag col / C-tile col
  const int hi = lane >> 4;

  const float* hpB = H + ((size_t)b * NNODE + nb) * NH;
  const float* agB = agg + ((size_t)b * NNODE + nb) * NH;
  const float* xB = xbase + (((size_t)b * x_tstride + x_t) * NNODE + nb) * NF;
  float* hnB = (float*)hpB;  // in-place update (tile-private rows)
  float* outB = out + (((size_t)b * TOUT + out_t) * NNODE + nb) * NF;
  float* S = sScr + wave * 16 * SSTR;

  v16bf hA0 = afrag_g(hpB, NH, 0, m, hi);
  v16bf hA1 = afrag_g(hpB, NH, 32, m, hi);
  v16bf gA0 = afrag_g(agB, NH, 0, m, hi);
  v16bf gA1 = afrag_g(agB, NH, 32, m, hi);
  v16bf xA = afrag_g(xB, NF, 0, m, hi);

  // pre + tanh + residual, write Hnew (global + LDS scratch for transpose)
#pragma unroll
  for (int jt = 0; jt < 4; ++jt) {
    v8f acc;
    float dj = sD[jt * 16 + m];
#pragma unroll
    for (int r = 0; r < 8; ++r) acc[r] = dj;
    acc = wmma_bf16(hA0, bfrag_s(sA, NH, jt * 16, 0, m, hi), acc);
    acc = wmma_bf16(hA1, bfrag_s(sA, NH, jt * 16, 32, m, hi), acc);
    acc = wmma_bf16(gA0, bfrag_s(sC, NH, jt * 16, 0, m, hi), acc);
    acc = wmma_bf16(gA1, bfrag_s(sC, NH, jt * 16, 32, m, hi), acc);
    acc = wmma_bf16(xA, bfrag_s(sBm, NF, jt * 16, 0, m, hi), acc);
#pragma unroll
    for (int r = 0; r < 8; ++r) {       // C/D layout: row = r+8*hi, col = jt*16+m
      int row = r + 8 * hi;
      int col = jt * 16 + m;
      float hn = hpB[row * NH + col] + tanhf(acc[r]);
      hnB[row * NH + col] = hn;
      S[row * SSTR + col] = hn;
    }
  }

  // MLP layer 1 (same-wave LDS ops are in-order: no barrier needed)
  v16bf nA0 = afrag_s(S, 0, m, hi);
  v16bf nA1 = afrag_s(S, 32, m, hi);
#pragma unroll
  for (int jt = 0; jt < 4; ++jt) {
    v8f acc;
    float bj = sb1[jt * 16 + m];
#pragma unroll
    for (int r = 0; r < 8; ++r) acc[r] = bj;
    acc = wmma_bf16(nA0, bfrag_s(sW1, NH, jt * 16, 0, m, hi), acc);
    acc = wmma_bf16(nA1, bfrag_s(sW1, NH, jt * 16, 32, m, hi), acc);
#pragma unroll
    for (int r = 0; r < 8; ++r)
      S[(r + 8 * hi) * SSTR + jt * 16 + m] = fmaxf(acc[r], 0.0f);
  }

  // MLP layer 2
  nA0 = afrag_s(S, 0, m, hi);
  nA1 = afrag_s(S, 32, m, hi);
#pragma unroll
  for (int jt = 0; jt < 4; ++jt) {
    v8f acc;
    float bj = sb2[jt * 16 + m];
#pragma unroll
    for (int r = 0; r < 8; ++r) acc[r] = bj;
    acc = wmma_bf16(nA0, bfrag_s(sW2, NH, jt * 16, 0, m, hi), acc);
    acc = wmma_bf16(nA1, bfrag_s(sW2, NH, jt * 16, 32, m, hi), acc);
#pragma unroll
    for (int r = 0; r < 8; ++r)
      S[(r + 8 * hi) * SSTR + jt * 16 + m] = fmaxf(acc[r], 0.0f);
  }

  // Output layer: h2@W3^T + b3 + x@Fm^T + G  (16 x 32)
  nA0 = afrag_s(S, 0, m, hi);
  nA1 = afrag_s(S, 32, m, hi);
#pragma unroll
  for (int jt = 0; jt < 2; ++jt) {
    v8f acc;
    float bj = sb3[jt * 16 + m] + sG[jt * 16 + m];
#pragma unroll
    for (int r = 0; r < 8; ++r) acc[r] = bj;
    acc = wmma_bf16(nA0, bfrag_s(sW3, NH, jt * 16, 0, m, hi), acc);
    acc = wmma_bf16(nA1, bfrag_s(sW3, NH, jt * 16, 32, m, hi), acc);
    acc = wmma_bf16(xA, bfrag_s(sFm, NF, jt * 16, 0, m, hi), acc);
#pragma unroll
    for (int r = 0; r < 8; ++r)
      outB[(r + 8 * hi) * NF + jt * 16 + m] = acc[r];
  }
}

// ---------------------------------------------------------------------------
extern "C" void kernel_launch(void* const* d_in, const int* in_sizes, int n_in,
                              void* d_out, int out_size, void* d_ws,
                              size_t ws_size, hipStream_t stream) {
  const float* x_in = (const float*)d_in[0];
  const int* esrc = (const int*)d_in[1];
  const int* edst = (const int*)d_in[2];
  const float* efeat = (const float*)d_in[3];
  const float* initH = (const float*)d_in[4];
  const float* A = (const float*)d_in[5];
  const float* Bm = (const float*)d_in[6];
  const float* C = (const float*)d_in[7];
  const float* Dv = (const float*)d_in[8];
  const float* Fm = (const float*)d_in[9];
  const float* G = (const float*)d_in[10];
  const float* W1 = (const float*)d_in[11];
  const float* b1 = (const float*)d_in[12];
  const float* W2 = (const float*)d_in[13];
  const float* b2 = (const float*)d_in[14];
  const float* W3 = (const float*)d_in[15];
  const float* b3 = (const float*)d_in[16];
  float* out = (float*)d_out;

  const size_t Hn = (size_t)NBB * NNODE * NH;  // 6.4M floats
  float* Hbuf = (float*)d_ws;                  // 25.6 MB
  float* Agg = Hbuf + Hn;                      // 25.6 MB

  fill_H0<<<(int)((Hn + 255) / 256), 256, 0, stream>>>(initH, Hbuf, Hn);

  const int tiles = NBB * (NNODE / 16);
  const int nodeBlocks = (tiles + WAVES - 1) / WAVES;
  const long long edgeThreads = (long long)NBB * NEDGE * 32;
  const int edgeBlocks = (int)((edgeThreads + 255) / 256);

  for (int t = 0; t < TOUT; ++t) {
    int ef_t = (t < NT) ? t : (NT - 1);
    zero_buf<<<(int)((Hn + 255) / 256), 256, 0, stream>>>(Agg, Hn);
    edge_scatter<<<edgeBlocks, 256, 0, stream>>>(Hbuf, esrc, edst, efeat, ef_t,
                                                 Agg);
    const float* xb;
    int xs, xt;
    if (t < NT) { xb = x_in; xs = NT; xt = t; }
    else        { xb = out;  xs = TOUT; xt = NT - 1; }  // x_last = xs[:, T-1]
    node_step<<<nodeBlocks, WAVES * 32, 0, stream>>>(
        Hbuf, Agg, xb, xs, xt, out, t, A, Bm, C, Dv, Fm, G, W1, b1, W2, b2, W3,
        b3);
  }
}
